// MoE_20409684591293
// MI455X (gfx1250) — matrix operand order repaired and checked by reading, not compiled, not run on hardware
//
#include <hip/hip_runtime.h>
#include <hip/hip_bf16.h>

// ---------------- problem constants ----------------
constexpr int kD   = 1024;   // model dim
constexpr int kH   = 4096;   // hidden dim
constexpr int kE   = 8;      // experts
constexpr int kT   = 4096;   // tokens = B*S
constexpr int kM   = 32;     // token tile (rows)
constexpr int kHC  = 128;    // h-chunk width
constexpr int kKC  = 128;    // k sub-chunk for GEMM1 staging
constexpr int kK2  = 32;     // k slice for GEMM2 staging
constexpr int kNP  = 512;    // GEMM2 n-phase width (2 phases of 512)

// LDS layout (bf16 elements), padded for conflict-free b128 fragment reads.
// Total 150,528 B < 160 KB so two workgroups can co-reside on a 320 KB WGP.
constexpr int XSTR  = kD  + 8;   // 1032
constexpr int W1STR = kKC + 8;   // 136
constexpr int W2STR = kK2 + 8;   // 40
constexpr int HSTR  = kHC + 8;   // 136
constexpr size_t OFF_X  = 0;
constexpr size_t OFF_W1 = OFF_X  + (size_t)kM  * XSTR  * 2;  // 66048
constexpr size_t OFF_W2 = OFF_W1 + (size_t)kKC * W1STR * 2;  // 100864
constexpr size_t OFF_H  = OFF_W2 + (size_t)kNP * W2STR * 2;  // 141824
constexpr size_t SMEM_BYTES = OFF_H + (size_t)kM * HSTR * 2; // 150528

typedef __bf16 bf16;
typedef __attribute__((ext_vector_type(4)))  bf16  v4bf;
typedef __attribute__((ext_vector_type(8)))  bf16  v8bf;
typedef __attribute__((ext_vector_type(16))) bf16  v16bf;
typedef __attribute__((ext_vector_type(8)))  float v8f;

// A-matrix 16x32 bf16 fragment (ISA 7.12.2): half = lane>>4,
//   elems 0..7  -> k = k0 + 8*half + {0..7}
//   elems 8..15 -> k = k0 + 16 + 8*half + {0..7}
__device__ __forceinline__ v16bf load_fragA(const bf16* __restrict__ row, int k0, int half) {
  const v8bf lo = *reinterpret_cast<const v8bf*>(row + k0 + 8 * half);
  const v8bf hi = *reinterpret_cast<const v8bf*>(row + k0 + 16 + 8 * half);
  return __builtin_shufflevector(lo, hi, 0,1,2,3,4,5,6,7,8,9,10,11,12,13,14,15);
}
// B-matrix 32x16 bf16 fragment (same k order as A):
//   elems 0..7 -> k = k0 + 8*half + {0..7}, elems 8..15 -> k = k0 + 16 + 8*half + {0..7}
__device__ __forceinline__ v16bf load_fragB(const bf16* __restrict__ col, int k0, int half) {
  const v8bf lo = *reinterpret_cast<const v8bf*>(col + k0 + 8 * half);
  const v8bf hi = *reinterpret_cast<const v8bf*>(col + k0 + 16 + 8 * half);
  return __builtin_shufflevector(lo, hi, 0,1,2,3,4,5,6,7,8,9,10,11,12,13,14,15);
}

// ---------------- kernel 1: gating + top-2 + softmax ----------------
__global__ void __launch_bounds__(128)
gate_topk(const float* __restrict__ x, const float* __restrict__ gw,
          const float* __restrict__ gb, int* __restrict__ topi,
          float* __restrict__ topw) {
  const int lane = threadIdx.x & 31;
  const int wid  = threadIdx.x >> 5;
  const int t    = blockIdx.x * 4 + wid;
  const float* xr = x + (size_t)t * kD;

  float acc[kE];
#pragma unroll
  for (int e = 0; e < kE; ++e) acc[e] = 0.f;

  for (int d = lane; d < kD; d += 32) {
    const float xv = xr[d];
    const float4* g4 = reinterpret_cast<const float4*>(gw + (size_t)d * kE);
    const float4 a = g4[0], b = g4[1];
    acc[0] += xv * a.x; acc[1] += xv * a.y; acc[2] += xv * a.z; acc[3] += xv * a.w;
    acc[4] += xv * b.x; acc[5] += xv * b.y; acc[6] += xv * b.z; acc[7] += xv * b.w;
  }
#pragma unroll
  for (int off = 16; off > 0; off >>= 1)
#pragma unroll
    for (int e = 0; e < kE; ++e) acc[e] += __shfl_xor(acc[e], off, 32);

  if (lane == 0) {
    float v[kE];
#pragma unroll
    for (int e = 0; e < kE; ++e) v[e] = acc[e] + gb[e];
    int i0 = 0; float b0 = v[0];
#pragma unroll
    for (int e = 1; e < kE; ++e) if (v[e] > b0) { b0 = v[e]; i0 = e; }   // first-index ties
    int i1 = -1; float b1v = -3.4e38f;
#pragma unroll
    for (int e = 0; e < kE; ++e)
      if (e != i0 && v[e] > b1v) { b1v = v[e]; i1 = e; }
    const float e1 = __expf(b1v - b0);            // stable: b1v <= b0
    const float inv = 1.f / (1.f + e1);
    topi[t * 2 + 0] = i0;  topi[t * 2 + 1] = i1;
    topw[t * 2 + 0] = inv; topw[t * 2 + 1] = e1 * inv;
  }
}

// ---------------- kernel 2: group tokens into (expert, slot) segments ----------------
__global__ void build_lists(const int* __restrict__ topi, const float* __restrict__ topw,
                            int* __restrict__ counts, int* __restrict__ lists,
                            float* __restrict__ wts) {
  const int s = threadIdx.x;        // 0..15 : seg = expert*2 + slot
  if (s >= 2 * kE) return;
  const int e = s >> 1, slot = s & 1;
  int cnt = 0;
  for (int t = 0; t < kT; ++t) {
    if (topi[t * 2 + slot] == e) {
      lists[(size_t)s * kT + cnt] = t;
      wts[(size_t)s * kT + cnt]   = topw[t * 2 + slot];
      ++cnt;
    }
  }
  counts[s] = cnt;
}

// ---------------- kernel 3: fused grouped expert FFN ----------------
// Block = 8 waves, 32 tokens of one (expert, slot) segment, full 1024-wide output.
// Epilogue: 2 atomic f32 adds per output element (one per slot) onto exact 0.0;
// fp add is commutative -> bitwise deterministic regardless of scheduling.
__global__ void __launch_bounds__(256)
moe_ffn(const float* __restrict__ x,  const float* __restrict__ w1,
        const float* __restrict__ b1, const float* __restrict__ w2,
        const float* __restrict__ b2, const int* __restrict__ counts,
        const int* __restrict__ lists, const float* __restrict__ wts,
        float* __restrict__ out) {
  constexpr int NTILE = kT / kM;            // 128 tiles max per segment
  const int seg  = blockIdx.x / NTILE;      // 0..15 = expert*2 + slot
  const int tile = blockIdx.x % NTILE;
  const int e    = seg >> 1;
  const int cnt  = counts[seg];
  const int base = tile * kM;
  if (base >= cnt) return;                  // uniform early-exit, before any barrier
  const int valid = min(kM, cnt - base);

  extern __shared__ char smem[];
  bf16* sX  = reinterpret_cast<bf16*>(smem + OFF_X);   // [32][XSTR]
  bf16* sW1 = reinterpret_cast<bf16*>(smem + OFF_W1);  // [128][W1STR] (h-major, transposed)
  bf16* sW2 = reinterpret_cast<bf16*>(smem + OFF_W2);  // [512][W2STR] (n-major, transposed)
  bf16* sH  = reinterpret_cast<bf16*>(smem + OFF_H);   // [32][HSTR]
  __shared__ int   s_tok[kM];
  __shared__ float s_wt[kM];

  const int tid = threadIdx.x;
  if (tid < kM) {
    const int i = base + tid;
    if (i < cnt) { s_tok[tid] = lists[(size_t)seg * kT + i]; s_wt[tid] = wts[(size_t)seg * kT + i]; }
    else         { s_tok[tid] = lists[(size_t)seg * kT + base]; s_wt[tid] = 0.f; }
  }
  __syncthreads();

  // Stage x tile: 32 tokens x 1024, fp32 -> bf16 (float4 reads, packed bf16x4 stores)
  for (int idx = tid; idx < kM * (kD / 4); idx += 256) {
    const int r = idx >> 8, c4 = (idx & 255) * 4;
    const float4 v = *reinterpret_cast<const float4*>(x + (size_t)s_tok[r] * kD + c4);
    v4bf p; p.x = (bf16)v.x; p.y = (bf16)v.y; p.z = (bf16)v.z; p.w = (bf16)v.w;
    *reinterpret_cast<v4bf*>(sX + r * XSTR + c4) = p;
  }
  __syncthreads();

  const int lane = tid & 31, wid = tid >> 5;
  const int l15 = lane & 15, half = lane >> 4;

  v8f acc[2][8];                            // [row-half][phase*4 + j]
#pragma unroll
  for (int p = 0; p < 2; ++p)
#pragma unroll
    for (int i = 0; i < 8; ++i) acc[p][i] = v8f{};

  const float* w1e = w1 + (size_t)e * kD * kH;
  const float* w2e = w2 + (size_t)e * kH * kD;

  for (int hc = 0; hc < kH; hc += kHC) {
    // ---- GEMM1: h[32,128] = x[32,1024] @ w1[:, hc:hc+128] ----
    v8f hacc[2] = {v8f{}, v8f{}};
    for (int k0 = 0; k0 < kD; k0 += kKC) {
      for (int idx = tid; idx < kKC * (kHC / 4); idx += 256) {  // stage transposed bf16
        const int kk = idx >> 5, h4 = (idx & 31) * 4;
        const float4 v = *reinterpret_cast<const float4*>(w1e + (size_t)(k0 + kk) * kH + hc + h4);
        sW1[(h4 + 0) * W1STR + kk] = (bf16)v.x;
        sW1[(h4 + 1) * W1STR + kk] = (bf16)v.y;
        sW1[(h4 + 2) * W1STR + kk] = (bf16)v.z;
        sW1[(h4 + 3) * W1STR + kk] = (bf16)v.w;
      }
      if (k0 + kKC < kD)
        __builtin_prefetch(w1e + (size_t)(k0 + kKC) * kH + hc, 0, 1);
      __syncthreads();
      const bf16* arow0 = sX + l15 * XSTR;
      const bf16* arow1 = sX + (16 + l15) * XSTR;
      const bf16* bcol  = sW1 + (wid * 16 + l15) * W1STR;
#pragma unroll
      for (int ks = 0; ks < kKC; ks += 32) {
        const v16bf b  = load_fragB(bcol, ks, half);
        const v16bf a0 = load_fragA(arow0, k0 + ks, half);
        const v16bf a1 = load_fragA(arow1, k0 + ks, half);
        hacc[0] = __builtin_amdgcn_wmma_f32_16x16x32_bf16(false, a0, false, b,
                                                          (short)0, hacc[0], false, false);
        hacc[1] = __builtin_amdgcn_wmma_f32_16x16x32_bf16(false, a1, false, b,
                                                          (short)0, hacc[1], false, false);
      }
      __syncthreads();
    }
    // ---- bias + exact GELU, h -> LDS bf16 ----
    {
      const int cl = wid * 16 + l15;
      const float bb = b1[(size_t)e * kH + hc + cl];
#pragma unroll
      for (int p = 0; p < 2; ++p)
#pragma unroll
        for (int r = 0; r < 8; ++r) {
          const float v = hacc[p][r] + bb;
          const float g = 0.5f * v * (1.0f + erff(v * 0.70710678118654752f));
          sH[(p * 16 + r + 8 * half) * HSTR + cl] = (bf16)g;
        }
    }
    __syncthreads();
    // ---- GEMM2: y[32,1024] += h[32,128] @ w2[hc:hc+128, :]  (two 512-col phases) ----
    for (int kk = 0; kk < kHC; kk += kK2) {
      const v16bf a0 = load_fragA(sH + l15 * HSTR, kk, half);       // live across phases
      const v16bf a1 = load_fragA(sH + (16 + l15) * HSTR, kk, half);
#pragma unroll
      for (int ph = 0; ph < 2; ++ph) {
        for (int idx = tid; idx < kK2 * (kNP / 4); idx += 256) {    // stage transposed bf16
          const int kl = idx >> 7, n4 = (idx & 127) * 4;
          const float4 v = *reinterpret_cast<const float4*>(
              w2e + (size_t)(hc + kk + kl) * kD + ph * kNP + n4);
          sW2[(n4 + 0) * W2STR + kl] = (bf16)v.x;
          sW2[(n4 + 1) * W2STR + kl] = (bf16)v.y;
          sW2[(n4 + 2) * W2STR + kl] = (bf16)v.z;
          sW2[(n4 + 3) * W2STR + kl] = (bf16)v.w;
        }
        if (ph == 0)
          __builtin_prefetch(w2e + (size_t)(hc + kk) * kD + kNP, 0, 1);
        else if (kk + kK2 < kHC)
          __builtin_prefetch(w2e + (size_t)(hc + kk + kK2) * kD, 0, 1);
        __syncthreads();
#pragma unroll
        for (int j = 0; j < 4; ++j) {
          const bf16* bcol = sW2 + (wid * 64 + j * 16 + l15) * W2STR;
          const v16bf b = load_fragB(bcol, 0, half);
          acc[0][ph * 4 + j] = __builtin_amdgcn_wmma_f32_16x16x32_bf16(
              false, a0, false, b, (short)0, acc[0][ph * 4 + j], false, false);
          acc[1][ph * 4 + j] = __builtin_amdgcn_wmma_f32_16x16x32_bf16(
              false, a1, false, b, (short)0, acc[1][ph * 4 + j], false, false);
        }
        __syncthreads();
      }
    }
  }

  // ---- epilogue: out[token] += prob * (y + b2)  (2 atomics/element total) ----
#pragma unroll
  for (int ph = 0; ph < 2; ++ph)
#pragma unroll
    for (int j = 0; j < 4; ++j) {
      const int col = ph * kNP + wid * 64 + j * 16 + l15;
      const float bb = b2[(size_t)e * kD + col];
#pragma unroll
      for (int p = 0; p < 2; ++p)
#pragma unroll
        for (int r = 0; r < 8; ++r) {
          const int row = p * 16 + r + 8 * half;
          if (row < valid) {
            const float v = s_wt[row] * (acc[p][ph * 4 + j][r] + bb);
            atomicAdd(out + (size_t)s_tok[row] * kD + col, v);
          }
        }
    }
}

// ---------------- launcher ----------------
extern "C" void kernel_launch(void* const* d_in, const int* in_sizes, int n_in,
                              void* d_out, int out_size, void* d_ws, size_t ws_size,
                              hipStream_t stream) {
  (void)in_sizes; (void)n_in; (void)ws_size;
  const float* x  = (const float*)d_in[0];
  const float* gw = (const float*)d_in[1];
  const float* gb = (const float*)d_in[2];
  const float* w1 = (const float*)d_in[3];
  const float* b1 = (const float*)d_in[4];
  const float* w2 = (const float*)d_in[5];
  const float* b2 = (const float*)d_in[6];
  float* out = (float*)d_out;

  char* ws = (char*)d_ws;
  int*   topi   = (int*)(ws);                                   // [T*2]
  float* topw   = (float*)(ws + (size_t)2 * kT * 4);            // [T*2]
  int*   counts = (int*)(ws + (size_t)4 * kT * 4);              // [16]
  int*   lists  = (int*)(ws + (size_t)4 * kT * 4 + 64);         // [16][T]
  float* wlist  = (float*)(ws + (size_t)4 * kT * 4 + 64 + (size_t)16 * kT * 4);

  hipMemsetAsync(out, 0, (size_t)out_size * sizeof(float), stream);
  gate_topk<<<kT / 4, 128, 0, stream>>>(x, gw, gb, topi, topw);
  build_lists<<<1, 16, 0, stream>>>(topi, topw, counts, lists, wlist);

  const dim3 grid(2 * kE * (kT / kM));      // 16 segments x 128 tiles
  moe_ffn<<<grid, 256, SMEM_BYTES, stream>>>(x, w1, b1, w2, b2, counts, lists, wlist, out);
}
